// MultiHeadAttentionLayer_65154653880383
// MI455X (gfx1250) — matrix operand order repaired and checked by reading, not compiled, not run on hardware
//
#include <hip/hip_runtime.h>
#include <hip/hip_bf16.h>

#define IN_DIM   256
#define NUM_HEAD 8
#define OUT_DIM  32
#define DMODEL   256          // NUM_HEAD*OUT_DIM
#define QKVCOLS  768          // q|k|v concatenated per node

typedef __attribute__((ext_vector_type(16))) _Float16 v16h;
typedef __attribute__((ext_vector_type(8)))  _Float16 v8h;
typedef __attribute__((ext_vector_type(4)))  _Float16 v4h;
typedef __attribute__((ext_vector_type(8)))  float    v8f;
typedef __attribute__((ext_vector_type(4)))  unsigned int v4u;
typedef __attribute__((ext_vector_type(8)))  int      v8i;
typedef __attribute__((ext_vector_type(4)))  int      v4i;

#if defined(__has_builtin)
#if __has_builtin(__builtin_amdgcn_tensor_load_to_lds)
#define HAVE_TDM 1
#endif
#if __has_builtin(__builtin_amdgcn_s_wait_tensorcnt)
#define HAVE_WAIT_TENSOR 1
#endif
#endif

#ifdef HAVE_TDM
// ---------------------------------------------------------------------------
// Issue one TDM 2D tile load (2-byte elements) per ISA D# layout (cdna5 §8).
// Zero-filled groups 2/3 (2D tile: tile_dim2=0 marks higher dims unused).
// OOB rows (y >= tensor_d1) read as zero -> handles ragged last M tile.
// ---------------------------------------------------------------------------
static __device__ __forceinline__ void tdm_load_2d_f16(
    const void* gaddr, unsigned lds_off,
    unsigned tile_d0, unsigned tile_d1,
    unsigned tensor_d0, unsigned tensor_d1, unsigned stride0_elems)
{
    unsigned long long ga = (unsigned long long)(size_t)gaddr;
    v4u g0;
    g0[0] = 1u;                                       // count=1, user descriptor
    g0[1] = lds_off;                                  // lds_addr (bytes)
    g0[2] = (unsigned)(ga & 0xffffffffu);             // global_addr[31:0]
    g0[3] = (unsigned)((ga >> 32) & 0x01ffffffu)      // global_addr[56:32]
          | (2u << 30);                               // type=2 ("image")
    v8i g1;
    g1[0] = (int)(1u << 16);                          // data_size=1 (2 bytes)
    g1[1] = (int)((tensor_d0 & 0xffffu) << 16);       // tensor_dim0[15:0]
    g1[2] = (int)((tensor_d0 >> 16) | ((tensor_d1 & 0xffffu) << 16));
    g1[3] = (int)((tensor_d1 >> 16) | (tile_d0 << 16));
    g1[4] = (int)(tile_d1 & 0xffffu);                 // tile_dim1 (tile_dim2=0)
    g1[5] = (int)stride0_elems;                       // tensor_dim0_stride[31:0]
    g1[6] = 0;
    g1[7] = 0;
    v4i z4 = {0, 0, 0, 0};
#if __has_include(<hip/amd_detail/amd_gfx1250_TDM.h>)
    v8i z8 = {0, 0, 0, 0, 0, 0, 0, 0};
    __builtin_amdgcn_tensor_load_to_lds(g0, g1, z4, z4, z8, 0);
#else
    __builtin_amdgcn_tensor_load_to_lds(g0, g1, z4, z4, 0);
#endif
}

static __device__ __forceinline__ void tdm_wait()
{
#ifdef HAVE_WAIT_TENSOR
    __builtin_amdgcn_s_wait_tensorcnt(0);
#else
    asm volatile("s_wait_tensorcnt 0x0" ::: "memory");
#endif
}
#endif // HAVE_TDM

// ---------------------------------------------------------------------------
// Kernel 1a: h (f32 [N,256]) -> h16 (f16), one pass, float4 vectorized.
// ---------------------------------------------------------------------------
__global__ __launch_bounds__(256) void cvt_h_kernel(
    const float* __restrict__ h, _Float16* __restrict__ h16, int total4)
{
    int i4 = blockIdx.x * 256 + threadIdx.x;
    if (i4 >= total4) return;
    float4 x = *(const float4*)&h[(size_t)i4 * 4];
    v4h hx;
    hx[0] = (_Float16)x.x; hx[1] = (_Float16)x.y;
    hx[2] = (_Float16)x.z; hx[3] = (_Float16)x.w;
    *(v4h*)&h16[(size_t)i4 * 4] = hx;
}

// ---------------------------------------------------------------------------
// Kernel 1b: pack Wq|Wk|Wv into TRANSPOSED f16 matrix W16t [768 cols][256 K].
// ---------------------------------------------------------------------------
__global__ __launch_bounds__(256) void cvt_weights_kernel(
    const float* __restrict__ Wq, const float* __restrict__ Wk,
    const float* __restrict__ Wv, _Float16* __restrict__ W16t)
{
    int idx = blockIdx.x * 256 + threadIdx.x;      // 0 .. 768*256-1
    int n = idx >> 8;
    int k = idx & 255;
    float w;
    if (n < 256)      w = Wq[k * 256 + n];
    else if (n < 512) w = Wk[k * 256 + (n - 256)];
    else              w = Wv[k * 256 + (n - 512)];
    W16t[idx] = (_Float16)w;
}

// ---------------------------------------------------------------------------
// Kernel 1c: fused bias vector b_all[768] = bq|bk|bv (f32).
// ---------------------------------------------------------------------------
__global__ __launch_bounds__(256) void cvt_bias_kernel(
    const float* __restrict__ bq, const float* __restrict__ bk,
    const float* __restrict__ bv, float* __restrict__ b_all)
{
    int j = blockIdx.x * 256 + threadIdx.x;        // 0..767
    float b;
    if (j < 256)      b = bq[j];
    else if (j < 512) b = bk[j - 256];
    else              b = bv[j - 512];
    b_all[j] = b;
}

// ---------------------------------------------------------------------------
// Kernel 2: qkv[N,768] = h16 @ W + bias   via v_wmma_f32_16x16x32_f16.
// Block tile 128(M) x 64(N); 8 waves, 16-row strip per wave, 4 accumulators.
// B strip (64x256 f16, 32KB) staged once. A staged in 128x64 chunks,
// DOUBLE-BUFFERED: TDM for chunk kc+1 is issued before computing chunk kc,
// s_wait_tensorcnt deferred until after the 8 WMMAs (DMA overlaps compute).
// ---------------------------------------------------------------------------
__global__ __launch_bounds__(256) void qkv_gemm_kernel(
    const _Float16* __restrict__ h16, const _Float16* __restrict__ W16t,
    const float* __restrict__ b_all, float* __restrict__ qkv, int N)
{
    __shared__ _Float16 As[2][128 * 64];  // double-buffered A, 2 x 16 KB
    __shared__ _Float16 Bs[64 * 256];     // [n][k] full K strip, 32 KB

    const int t    = threadIdx.x;
    const int wave = t >> 5;
    const int lane = t & 31;
    const int m0   = blockIdx.y * 128;
    const int n0   = blockIdx.x * 64;

    v8f c[4] = {};

    // fragment coordinates (ISA 7.12.2 layouts)
    const int ar   = (wave << 4) + (lane & 15);
    const int k8   = (lane >> 4) << 3;
    const int bk16 = (lane >> 4) << 4;

#ifdef HAVE_TDM
    const unsigned asOff0 = (unsigned)(size_t)(void*)&As[0][0];
    const unsigned asOff1 = (unsigned)(size_t)(void*)&As[1][0];
    const unsigned bsOff  = (unsigned)(size_t)(void*)&Bs[0];
    if (wave == 0) {
        // B strip: 64 rows x 256 K from W16t[768,256]
        tdm_load_2d_f16(&W16t[(size_t)n0 * 256], bsOff,
                        256, 64, 256, (unsigned)(768 - n0), 256);
        // A chunk kc=0 into buffer 0
        tdm_load_2d_f16(&h16[(size_t)m0 * 256], asOff0,
                        64, 128, 64, (unsigned)(N - m0), 256);
        tdm_wait();
    }
    __syncthreads();
#else
    // ---- manual B strip staging ----
    #pragma unroll
    for (int i = 0; i < 8; ++i) {
        int idx8 = i * 2048 + t * 8;
        int n = idx8 >> 8;
        int k = idx8 & 255;
        *(v8h*)&Bs[idx8] = *(const v8h*)&W16t[(size_t)(n0 + n) * 256 + k];
    }
    // ---- manual A chunk kc=0 into buffer 0 ----
    #pragma unroll
    for (int i = 0; i < 4; ++i) {
        int idx8 = i * 2048 + t * 8;
        int r   = idx8 >> 6;
        int col = idx8 & 63;
        int grow = m0 + r;
        if (grow >= N) grow = N - 1;
        *(v8h*)&As[0][idx8] = *(const v8h*)&h16[(size_t)grow * IN_DIM + col];
    }
    __syncthreads();
#endif

    for (int kc = 0; kc < 4; ++kc) {
        const int k0 = kc * 64;
        const _Float16* Ac = As[kc & 1];

        // ---- kick off staging of the NEXT chunk into the other buffer ----
        if (kc < 3) {
#ifdef HAVE_TDM
            if (wave == 0) {
                tdm_load_2d_f16(&h16[(size_t)m0 * 256 + (kc + 1) * 64],
                                (kc & 1) ? asOff0 : asOff1,
                                64, 128, 64, (unsigned)(N - m0), 256);
            }
#else
            const int knext = (kc + 1) * 64;
            _Float16* An = As[(kc + 1) & 1];
            #pragma unroll
            for (int i = 0; i < 4; ++i) {
                int idx8 = i * 2048 + t * 8;
                int r   = idx8 >> 6;
                int col = idx8 & 63;
                int grow = m0 + r;
                if (grow >= N) grow = N - 1;
                *(v8h*)&An[idx8] = *(const v8h*)&h16[(size_t)grow * IN_DIM + knext + col];
            }
#endif
        }

        // ---- compute 8 WMMAs from the current buffer ----
        #pragma unroll
        for (int ki = 0; ki < 64; ki += 32) {
            union { v16h v; v8h h2[2]; } a;
            a.h2[0] = *(const v8h*)&Ac[ar * 64 + ki + k8];
            a.h2[1] = *(const v8h*)&Ac[ar * 64 + ki + 16 + k8];
            #pragma unroll
            for (int ns = 0; ns < 4; ++ns) {
                int bn = (ns << 4) + (lane & 15);
                union { v16h v; v8h h2[2]; } bb;
                bb.h2[0] = *(const v8h*)&Bs[bn * 256 + k0 + ki + k8];
                bb.h2[1] = *(const v8h*)&Bs[bn * 256 + k0 + ki + 16 + k8];
                v16h b = bb.v;
                c[ns] = __builtin_amdgcn_wmma_f32_16x16x32_f16(
                            false, a.v, false, b, (short)0, c[ns], false, false);
            }
        }

        if (kc < 3) {
#ifdef HAVE_TDM
            if (wave == 0) tdm_wait();      // deferred: DMA overlapped compute
#endif
            __syncthreads();
        }
    }

    // ---- epilogue: lanes 0-15 -> rows 0-7, lanes 16-31 -> rows 8-15 ----
    const int rbase = (lane >> 4) << 3;
    const int ncol  = lane & 15;
    #pragma unroll
    for (int ns = 0; ns < 4; ++ns) {
        int col = n0 + (ns << 4) + ncol;
        float bias = b_all[col];
        #pragma unroll
        for (int r = 0; r < 8; ++r) {
            int row = m0 + (wave << 4) + rbase + r;
            if (row < N)
                qkv[(size_t)row * QKVCOLS + col] = c[ns][r] + bias;
        }
    }
}

// ---------------------------------------------------------------------------
// Kernel 3: per-edge attention. One wave per edge; 4 lanes per head, 8 dims
// per lane. qkv stays L2-resident (153.6 MB < 192 MB L2).
// ---------------------------------------------------------------------------
__global__ __launch_bounds__(256) void edge_kernel(
    const float* __restrict__ qkv, const int* __restrict__ src,
    const int* __restrict__ dst, float* __restrict__ out,
    float* __restrict__ wZ, int E)
{
    int lane = threadIdx.x & 31;
    int e = blockIdx.x * 8 + (threadIdx.x >> 5);
    if (e >= E) return;

    int s = src[e];
    int d = dst[e];
    int head = lane >> 2;
    int sub  = (lane & 3) << 3;

    size_t qb = (size_t)d * QKVCOLS +       head * OUT_DIM + sub;
    size_t kb = (size_t)s * QKVCOLS + 256 + head * OUT_DIM + sub;
    size_t vb = (size_t)s * QKVCOLS + 512 + head * OUT_DIM + sub;

    float4 q0 = *(const float4*)(qkv + qb);
    float4 q1 = *(const float4*)(qkv + qb + 4);
    float4 k0 = *(const float4*)(qkv + kb);
    float4 k1 = *(const float4*)(qkv + kb + 4);

    float p = q0.x * k0.x + q0.y * k0.y + q0.z * k0.z + q0.w * k0.w
            + q1.x * k1.x + q1.y * k1.y + q1.z * k1.z + q1.w * k1.w;
    p += __shfl_xor(p, 1);
    p += __shfl_xor(p, 2);

    float score = __expf(p * 0.17677669529663687f);   // 1/sqrt(32)

    float4 v0 = *(const float4*)(qkv + vb);
    float4 v1 = *(const float4*)(qkv + vb + 4);

    float* o = out + (size_t)d * DMODEL + head * OUT_DIM + sub;
    atomicAdd(o + 0, score * v0.x);
    atomicAdd(o + 1, score * v0.y);
    atomicAdd(o + 2, score * v0.z);
    atomicAdd(o + 3, score * v0.w);
    atomicAdd(o + 4, score * v1.x);
    atomicAdd(o + 5, score * v1.y);
    atomicAdd(o + 6, score * v1.z);
    atomicAdd(o + 7, score * v1.w);
    if ((lane & 3) == 0)
        atomicAdd(wZ + (size_t)d * NUM_HEAD + head, score);
}

// ---------------------------------------------------------------------------
// Kernel 4: out[n,h,d] /= (wZ[n,h] + eps)
// ---------------------------------------------------------------------------
__global__ __launch_bounds__(256) void normalize_kernel(
    float* __restrict__ out, const float* __restrict__ wZ, int total)
{
    int idx = blockIdx.x * 256 + threadIdx.x;
    if (idx >= total) return;
    out[idx] = out[idx] / (wZ[idx >> 5] + 1e-9f);   // idx>>5 == n*8 + h
}

// ---------------------------------------------------------------------------
extern "C" void kernel_launch(void* const* d_in, const int* in_sizes, int n_in,
                              void* d_out, int out_size, void* d_ws, size_t ws_size,
                              hipStream_t stream)
{
    const float* h   = (const float*)d_in[0];
    const int*   src = (const int*)  d_in[1];
    const int*   dst = (const int*)  d_in[2];
    const float* Wq  = (const float*)d_in[3];
    const float* bq  = (const float*)d_in[4];
    const float* Wk  = (const float*)d_in[5];
    const float* bk  = (const float*)d_in[6];
    const float* Wv  = (const float*)d_in[7];
    const float* bv  = (const float*)d_in[8];
    float* out = (float*)d_out;

    const int N = in_sizes[0] / IN_DIM;
    const int E = in_sizes[1];

    // ws: qkv f32[N,768] | W16t f16[768,256] | h16 f16[N,256] | b_all f32[768] | wZ f32[N,8]
    char* ws = (char*)d_ws;
    float* qkv = (float*)ws;
    size_t off = (size_t)N * QKVCOLS * sizeof(float);
    _Float16* W16t = (_Float16*)(ws + off);
    off += (size_t)QKVCOLS * IN_DIM * sizeof(_Float16);
    _Float16* h16 = (_Float16*)(ws + off);
    off += (size_t)N * IN_DIM * sizeof(_Float16);
    float* b_all = (float*)(ws + off);
    off += (size_t)QKVCOLS * sizeof(float);
    float* wZ = (float*)(ws + off);

    // Zero accumulators every call (harness does not re-poison between replays)
    hipMemsetAsync(out, 0, (size_t)out_size * sizeof(float), stream);
    hipMemsetAsync(wZ, 0, (size_t)N * NUM_HEAD * sizeof(float), stream);

    // 1) precompute f16 operands + fused bias
    int total4 = (N * IN_DIM) / 4;
    cvt_h_kernel<<<(total4 + 255) / 256, 256, 0, stream>>>(h, h16, total4);
    cvt_weights_kernel<<<(QKVCOLS * IN_DIM) / 256, 256, 0, stream>>>(Wq, Wk, Wv, W16t);
    cvt_bias_kernel<<<QKVCOLS / 256, 256, 0, stream>>>(bq, bk, bv, b_all);

    // 2) fused QKV GEMM (WMMA + double-buffered TDM staging)
    dim3 ggrid(QKVCOLS / 64, (N + 127) / 128);
    qkv_gemm_kernel<<<ggrid, 256, 0, stream>>>(h16, W16t, b_all, qkv, N);

    // 3) per-edge attention scatter
    int eblocks = (E + 7) / 8;
    edge_kernel<<<eblocks, 256, 0, stream>>>(qkv, src, dst, out, wZ, E);

    // 4) normalize
    int total = N * DMODEL;
    normalize_kernel<<<(total + 255) / 256, 256, 0, stream>>>(out, wZ, total);
}
